// CustomGPT2Attention_80032420593940
// MI455X (gfx1250) — compile-verified, matrix operand order checked by reading, not run on hardware
//
#include <hip/hip_runtime.h>

#define BQ 2
#define SQ 2048
#define EQ 1024
#define HQ 16
#define DQ 64

typedef __attribute__((ext_vector_type(16))) __bf16 v16bf;
typedef __attribute__((ext_vector_type(8)))  float  v8f;
typedef unsigned int u32x4 __attribute__((ext_vector_type(4)));
typedef int          i32x8 __attribute__((ext_vector_type(8)));
typedef int          i32x4 __attribute__((ext_vector_type(4)));

union FragAB { v16bf v; unsigned short u[16]; };

__device__ __forceinline__ unsigned short f2bf(float f) {
    unsigned int u = __float_as_uint(f);
    u += 0x7FFFu + ((u >> 16) & 1u);          // round-to-nearest-even
    return (unsigned short)(u >> 16);
}
__device__ __forceinline__ float bf2f(unsigned short h) {
    return __uint_as_float(((unsigned int)h) << 16);
}

// DPP16 lane swap within a 16-lane row (VALU only, co-executes with WMMA)
template <int CTRL>
__device__ __forceinline__ float dppf(float x) {
    return __int_as_float(__builtin_amdgcn_update_dpp(
        0, __float_as_int(x), CTRL, 0xF, 0xF, true));
}
// max over 16 lanes: xor1(quad_perm 1,0,3,2) xor2(2,3,0,1) xor4(half_mirror) xor8(mirror)
__device__ __forceinline__ float rowmax16(float m) {
    m = fmaxf(m, dppf<0x0B1>(m));
    m = fmaxf(m, dppf<0x04E>(m));
    m = fmaxf(m, dppf<0x141>(m));
    m = fmaxf(m, dppf<0x140>(m));
    return m;
}

// ---------------------------------------------------------------------------
// TDM: async 2D bf16 tile load global -> LDS via Tensor Data Mover descriptor.
// pad_interval code: 3 -> pad after 16 dwords, 4 -> after 32 dwords; pad = 1 dword.
// ---------------------------------------------------------------------------
__device__ __forceinline__ void tdm_load_2d_bf16(
    const unsigned short* gptr, unsigned lds_byte_addr,
    unsigned tile0, unsigned tile1, unsigned stride0,
    unsigned pad_en, unsigned pad_interval)
{
    unsigned long long ga = (unsigned long long)(size_t)gptr;
    u32x4 g0;
    g0[0] = 1u;                                            // count=1 (valid user D#)
    g0[1] = lds_byte_addr;                                 // lds_addr
    g0[2] = (unsigned)ga;                                  // global_addr[31:0]
    g0[3] = (unsigned)((ga >> 32) & 0x1FFFFFFu) | (2u << 30); // addr[56:32] | type=2
    i32x8 g1;
    g1[0] = (int)((1u << 16) | (pad_en << 20) | (pad_interval << 22)); // data_size=2B
    g1[1] = (int)(tile0 << 16);     // tensor_dim0[15:0]
    g1[2] = (int)(tile1 << 16);     // tensor_dim0 hi | tensor_dim1[15:0]
    g1[3] = (int)(tile0 << 16);     // tensor_dim1 hi | tile_dim0
    g1[4] = (int)tile1;             // tile_dim1 | tile_dim2=0
    g1[5] = (int)stride0;           // tensor_dim0_stride[31:0]
    g1[6] = 0;
    g1[7] = 0;
    i32x4 z4 = {0, 0, 0, 0};
    i32x8 z8 = {0, 0, 0, 0, 0, 0, 0, 0};
    __builtin_amdgcn_tensor_load_to_lds(g0, g1, z4, z4, z8, 0);
}

// ---------------------------------------------------------------------------
// fp32 -> bf16 elementwise convert
// ---------------------------------------------------------------------------
__global__ void cvt_f32_bf16(const float* __restrict__ src,
                             unsigned short* __restrict__ dst, int n) {
    int i = blockIdx.x * blockDim.x + threadIdx.x;
    if (i < n) dst[i] = f2bf(src[i]);
}

// ---------------------------------------------------------------------------
// bf16 GEMM + bias:  C[M,N] = A[M,K] @ B[K,N] + bias[N]
// 256 threads = 8 waves; tile 128(M) x 64(N); K in 32-chunks.
// Double-buffered TDM pipeline: wave0 streams A tiles, wave1 streams B tiles;
// load of chunk k+1 overlaps WMMA compute of chunk k.
// ---------------------------------------------------------------------------
template <bool OUT_BF16>
__global__ __launch_bounds__(256) void gemm_bf16_wmma(
    const unsigned short* __restrict__ A,   // [M,K] bf16
    const unsigned short* __restrict__ Bm,  // [K,N] bf16
    const float* __restrict__ bias,         // [N]
    float* __restrict__ Cf,
    unsigned short* __restrict__ Cb,
    int M, int N, int K)
{
    __shared__ unsigned short lA[2][128][34];  // 17-dword rows: conflict-free
    __shared__ unsigned short lB[2][32][64];

    const int tid   = threadIdx.x;
    const int wave  = tid >> 5;
    const int lane  = tid & 31;
    const int l16   = lane & 15;
    const int hh    = lane >> 4;
    const int kbase = hh * 8;

    const int n0 = blockIdx.x * 64;
    const int m0 = blockIdx.y * 128;

    unsigned ldsA[2], ldsB[2];
    #pragma unroll
    for (int i = 0; i < 2; ++i) {
        ldsA[i] = (unsigned)(size_t)&lA[i][0][0];
        ldsB[i] = (unsigned)(size_t)&lB[i][0][0];
    }

    v8f acc[4];
    #pragma unroll
    for (int t = 0; t < 4; ++t)
        #pragma unroll
        for (int e = 0; e < 8; ++e) acc[t][e] = 0.0f;

    const int nChunks = K >> 5;
    // prologue: stream chunk 0
    if (wave == 0)
        tdm_load_2d_bf16(A + (size_t)m0 * K, ldsA[0], 32, 128, (unsigned)K, 1, 3);
    else if (wave == 1)
        tdm_load_2d_bf16(Bm + (size_t)0 * N + n0, ldsB[0], 64, 32, (unsigned)N, 0, 0);

    for (int kc = 0; kc < nChunks; ++kc) {
        const int cur = kc & 1;
        if (wave < 2) __builtin_amdgcn_s_wait_tensorcnt(0);  // my stream done
        __syncthreads();        // tiles(kc) visible; buf(kc-1) fully consumed
        if (kc + 1 < nChunks) { // stream kc+1 while computing kc
            if (wave == 0)
                tdm_load_2d_bf16(A + (size_t)m0 * K + (kc + 1) * 32,
                                 ldsA[cur ^ 1], 32, 128, (unsigned)K, 1, 3);
            else if (wave == 1)
                tdm_load_2d_bf16(Bm + (size_t)((kc + 1) * 32) * N + n0,
                                 ldsB[cur ^ 1], 64, 32, (unsigned)N, 0, 0);
        }

        FragAB af;
        {
            int row = wave * 16 + l16;
            #pragma unroll
            for (int e = 0; e < 16; ++e) {
                int k = (e < 8) ? (kbase + e) : (kbase + 8 + e);
                af.u[e] = lA[cur][row][k];
            }
        }
        #pragma unroll
        for (int t = 0; t < 4; ++t) {
            FragAB bfr;
            int n = t * 16 + l16;
            #pragma unroll
            for (int e = 0; e < 16; ++e) {
                int k = (e < 8) ? (kbase + e) : (kbase + 8 + e);
                bfr.u[e] = lB[cur][k][n];
            }
            acc[t] = __builtin_amdgcn_wmma_f32_16x16x32_bf16(
                false, af.v, false, bfr.v, (short)0, acc[t], false, false);
        }
    }

    #pragma unroll
    for (int t = 0; t < 4; ++t) {
        int n = n0 + t * 16 + l16;
        float bv = bias[n];
        #pragma unroll
        for (int r = 0; r < 8; ++r) {
            int m = m0 + wave * 16 + r + hh * 8;
            float v = acc[t][r] + bv;
            if (OUT_BF16) Cb[(size_t)m * N + n] = f2bf(v);
            else          Cf[(size_t)m * N + n] = v;
        }
    }
}

// ---------------------------------------------------------------------------
// Flash attention (non-causal): 1 block = 4 waves = 64 q rows of one (b,h).
// Double-buffered TDM K/V streams (wave0=K, wave1=V); DPP row-max;
// row-sum via WMMA(P, ones).
// ---------------------------------------------------------------------------
__global__ __launch_bounds__(128) void attn_wmma(
    const unsigned short* __restrict__ qkv,   // bf16 [B*S, 3E]
    unsigned short* __restrict__ outbf)       // bf16 [B*S, E]
{
    __shared__ unsigned short lK[2][32][66];  // 33-dword rows: conflict-free
    __shared__ unsigned short lV[2][32][66];
    __shared__ unsigned short lP[4][16][32];

    const int tid   = threadIdx.x;
    const int wave  = tid >> 5;
    const int lane  = tid & 31;
    const int l16   = lane & 15;
    const int hh    = lane >> 4;
    const int kbase = hh * 8;

    const int qb = blockIdx.x & 31;           // S/64 = 32 q-blocks
    const int bh = blockIdx.x >> 5;
    const int b  = bh >> 4;
    const int h  = bh & 15;

    const int    q0        = qb * 64 + wave * 16;
    const size_t rowstride = 3 * EQ;
    const size_t baseQ     = (size_t)b * SQ * rowstride + (size_t)h * DQ;
    const unsigned short* gKbase = qkv + baseQ + EQ;
    const unsigned short* gVbase = qkv + baseQ + 2 * EQ;

    unsigned ldsK[2], ldsV[2];
    #pragma unroll
    for (int i = 0; i < 2; ++i) {
        ldsK[i] = (unsigned)(size_t)&lK[i][0][0];
        ldsV[i] = (unsigned)(size_t)&lV[i][0][0];
    }

    // prologue: stream key-tile 0 (overlaps with Q frag loads below)
    if (wave == 0)
        tdm_load_2d_bf16(gKbase, ldsK[0], 64, 32, (unsigned)rowstride, 1, 4);
    else if (wave == 1)
        tdm_load_2d_bf16(gVbase, ldsV[0], 64, 32, (unsigned)rowstride, 1, 4);

    // Q A-frags with 1/sqrt(D)=0.125 folded in (exact bf16 exponent shift)
    FragAB qa[2];
    {
        size_t qrow = baseQ + (size_t)(q0 + l16) * rowstride;
        #pragma unroll
        for (int c = 0; c < 2; ++c)
            #pragma unroll
            for (int e = 0; e < 16; ++e) {
                int d = c * 32 + ((e < 8) ? (kbase + e) : (kbase + 8 + e));
                qa[c].u[e] = f2bf(0.125f * bf2f(qkv[qrow + d]));
            }
    }
    FragAB ones;                              // B = all 1.0 for row-sum WMMA
    #pragma unroll
    for (int e = 0; e < 16; ++e) ones.u[e] = 0x3F80;

    v8f o[4];
    #pragma unroll
    for (int t = 0; t < 4; ++t)
        #pragma unroll
        for (int e = 0; e < 8; ++e) o[t][e] = 0.0f;

    float rmax[8], rsum[8];
    #pragma unroll
    for (int r = 0; r < 8; ++r) { rmax[r] = -3.0e38f; rsum[r] = 0.0f; }

    for (int kt = 0; kt < SQ / 32; ++kt) {
        const int cur = kt & 1;
        if (wave < 2) __builtin_amdgcn_s_wait_tensorcnt(0);
        __syncthreads();        // tiles(kt) visible; buf(kt-1) fully consumed
        if (kt + 1 < SQ / 32) { // stream kt+1 while computing kt
            size_t goff = (size_t)((kt + 1) * 32) * rowstride;
            if (wave == 0)
                tdm_load_2d_bf16(gKbase + goff, ldsK[cur ^ 1],
                                 64, 32, (unsigned)rowstride, 1, 4);
            else if (wave == 1)
                tdm_load_2d_bf16(gVbase + goff, ldsV[cur ^ 1],
                                 64, 32, (unsigned)rowstride, 1, 4);
        }

        // logits S = (Q*scale) @ K^T : 2 N-tiles x 2 chained k-steps
        v8f s[2];
        #pragma unroll
        for (int t = 0; t < 2; ++t) {
            #pragma unroll
            for (int e = 0; e < 8; ++e) s[t][e] = 0.0f;
            #pragma unroll
            for (int c = 0; c < 2; ++c) {
                FragAB kb;
                int key = t * 16 + l16;
                #pragma unroll
                for (int e = 0; e < 16; ++e) {
                    int d = c * 32 + ((e < 8) ? (kbase + e) : (kbase + 8 + e));
                    kb.u[e] = lK[cur][key][d];
                }
                s[t] = __builtin_amdgcn_wmma_f32_16x16x32_bf16(
                    false, qa[c].v, false, kb.v, (short)0, s[t], false, false);
            }
        }

        // online softmax: DPP row-max, exp, rescale O
        float alpha[8], p0[8], p1[8];
        #pragma unroll
        for (int r = 0; r < 8; ++r) {
            float m  = rowmax16(fmaxf(s[0][r], s[1][r]));
            float nm = fmaxf(rmax[r], m);
            alpha[r] = __expf(rmax[r] - nm);
            rmax[r]  = nm;
            p0[r] = __expf(s[0][r] - nm);
            p1[r] = __expf(s[1][r] - nm);
            #pragma unroll
            for (int t = 0; t < 4; ++t) o[t][r] *= alpha[r];
        }

        // P: C-layout -> A-layout via per-wave LDS staging
        #pragma unroll
        for (int r = 0; r < 8; ++r) {
            int m = r + hh * 8;
            lP[wave][m][l16]      = f2bf(p0[r]);
            lP[wave][m][16 + l16] = f2bf(p1[r]);
        }
        asm volatile("s_wait_dscnt 0" ::: "memory");   // in-wave LDS RAW
        FragAB pa;
        #pragma unroll
        for (int e = 0; e < 16; ++e) {
            int k = (e < 8) ? (kbase + e) : (kbase + 8 + e);
            pa.u[e] = lP[wave][l16][k];
        }

        // row-sum via WMMA: sums[r] = sum_k P[row][k] (lands in C-layout)
        v8f sums;
        #pragma unroll
        for (int e = 0; e < 8; ++e) sums[e] = 0.0f;
        sums = __builtin_amdgcn_wmma_f32_16x16x32_bf16(
            false, pa.v, false, ones.v, (short)0, sums, false, false);
        #pragma unroll
        for (int r = 0; r < 8; ++r) rsum[r] = rsum[r] * alpha[r] + sums[r];

        // O += P @ V : 4 d-tiles, 32-key K-dim per WMMA
        #pragma unroll
        for (int t = 0; t < 4; ++t) {
            FragAB vb;
            int d = t * 16 + l16;
            #pragma unroll
            for (int e = 0; e < 16; ++e) {
                int k = (e < 8) ? (kbase + e) : (kbase + 8 + e);
                vb.u[e] = lV[cur][k][d];
            }
            o[t] = __builtin_amdgcn_wmma_f32_16x16x32_bf16(
                false, pa.v, false, vb.v, (short)0, o[t], false, false);
        }
    }

    // normalize + merge heads -> bf16 [B*S, E]
    #pragma unroll
    for (int t = 0; t < 4; ++t)
        #pragma unroll
        for (int r = 0; r < 8; ++r) {
            int m   = q0 + r + hh * 8;
            int col = h * DQ + t * 16 + l16;
            float v = o[t][r] / rsum[r];
            outbf[((size_t)b * SQ + m) * EQ + col] = f2bf(v);
        }
}

// ---------------------------------------------------------------------------
extern "C" void kernel_launch(void* const* d_in, const int* in_sizes, int n_in,
                              void* d_out, int out_size, void* d_ws, size_t ws_size,
                              hipStream_t stream) {
    const float* x  = (const float*)d_in[0];   // hidden_states [B,S,E]
    const float* w1 = (const float*)d_in[1];   // c_attn_w [E,3E]
    const float* b1 = (const float*)d_in[2];   // c_attn_b [3E]
    const float* w2 = (const float*)d_in[3];   // c_proj_w [E,E]
    const float* b2 = (const float*)d_in[4];   // c_proj_b [E]
    float* out = (float*)d_out;

    char* ws = (char*)d_ws;
    size_t off = 0;
    auto alloc = [&](size_t bytes) {
        void* p = ws + off;
        off = (off + bytes + 255) & ~(size_t)255;
        return p;
    };
    unsigned short* xbf  = (unsigned short*)alloc((size_t)BQ * SQ * EQ * 2);
    unsigned short* w1bf = (unsigned short*)alloc((size_t)EQ * 3 * EQ * 2);
    unsigned short* w2bf = (unsigned short*)alloc((size_t)EQ * EQ * 2);
    unsigned short* qkv  = (unsigned short*)alloc((size_t)BQ * SQ * 3 * EQ * 2);
    unsigned short* att  = (unsigned short*)alloc((size_t)BQ * SQ * EQ * 2);
    (void)ws_size; (void)in_sizes; (void)n_in; (void)out_size;

    int n;
    n = BQ * SQ * EQ;  cvt_f32_bf16<<<(n + 255) / 256, 256, 0, stream>>>(x,  xbf,  n);
    n = EQ * 3 * EQ;   cvt_f32_bf16<<<(n + 255) / 256, 256, 0, stream>>>(w1, w1bf, n);
    n = EQ * EQ;       cvt_f32_bf16<<<(n + 255) / 256, 256, 0, stream>>>(w2, w2bf, n);

    dim3 g1(3 * EQ / 64, BQ * SQ / 128);
    gemm_bf16_wmma<true><<<g1, 256, 0, stream>>>(
        xbf, w1bf, b1, nullptr, qkv, BQ * SQ, 3 * EQ, EQ);

    attn_wmma<<<BQ * HQ * (SQ / 64), 128, 0, stream>>>(qkv, att);

    dim3 g2(EQ / 64, BQ * SQ / 128);
    gemm_bf16_wmma<false><<<g2, 256, 0, stream>>>(
        att, w2bf, b2, out, nullptr, BQ * SQ, EQ, EQ);
}
